// HPC_DV_67542655696947
// MI455X (gfx1250) — compile-verified, matrix operand order checked by reading, not run on hardware
//
#include <hip/hip_runtime.h>
#include <hip/hip_bf16.h>

typedef __attribute__((ext_vector_type(16))) _Float16 v16h;
typedef __attribute__((ext_vector_type(8)))  _Float16 v8h;
typedef __attribute__((ext_vector_type(8)))  float    v8f;

#define L_  2
#define BS_ 64
#define T_  512
#define N_  1024   // EC == CA1 == CA3

// Fast sigmoid: v_exp_f32 + v_rcp_f32 (no IEEE div sequence)
__device__ __forceinline__ float sigmoidf_(float x) {
    return __builtin_amdgcn_rcpf(1.0f + __expf(-x));
}

// -------------------------------------------------------------------------
// One-time: pack the three per-layer weight matrices into WMMA-B tile order
// (f16). Layout per (mat, l): [nTile(64)][kTile(32)][lane(32)][16 halves],
// matching v_wmma_f32_16x16x32_f16 B-operand striping (n = lane&15,
// k_local = (j&7) | ((lane>>4)&1)<<3 | (j>>3)<<4).
// -------------------------------------------------------------------------
__global__ __launch_bounds__(256) void pack_kernel(
    const float* __restrict__ wec3ca1, const float* __restrict__ wca1ec5,
    const float* __restrict__ inter_w, _Float16* __restrict__ packW)
{
    unsigned idx = blockIdx.x * 256u + threadIdx.x;      // < 3*2*1048576
    if (idx >= 6291456u) return;
    unsigned mat = idx >> 21;
    unsigned rem = idx & 2097151u;
    unsigned l   = rem >> 20;
    unsigned p   = rem & 1048575u;
    unsigned j     = p & 15u;
    unsigned lane  = (p >> 4) & 31u;
    unsigned kTile = (p >> 9) & 31u;
    unsigned nTile = p >> 14;
    unsigned nloc = lane & 15u;
    unsigned kloc = (j & 7u) | (((lane >> 4) & 1u) << 3) | ((j >> 3) << 4);
    unsigned k = kTile * 32u + kloc;
    unsigned n = nTile * 16u + nloc;
    float v;
    if (mat == 0)      v = wec3ca1[(l << 20) + k * 1024u + n];       // B[k][n]
    else if (mat == 1) v = wca1ec5[(l << 20) + k * 1024u + n];       // B[k][n]
    else               v = inter_w[(l << 20) + n * 1024u + k];       // Bᵀ
    packW[idx] = (_Float16)v;
}

// -------------------------------------------------------------------------
// One-time: s_ca3[t][l][c] = sigmoid(10*(gauss(t) @ wca3ca1[l])[c] - 5).
// Gaussian underflows f32 to exactly 0 beyond |j - 2t| ~ 145, so a +-150
// window is bit-identical to the full 1024-term sum in f32.
// -------------------------------------------------------------------------
__global__ __launch_bounds__(256) void sca3_kernel(
    const float* __restrict__ wca3ca1, float* __restrict__ sca3)
{
    unsigned idx = blockIdx.x * 256u + threadIdx.x;      // < 512*2*1024
    if (idx >= 1048576u) return;
    int t = idx >> 11;
    int l = (idx >> 10) & 1;
    int c = idx & 1023;
    float acc = 0.0f;
    const float ft = (float)t;
    const float step = 512.0f / 1023.0f;
    int j0 = 2 * t - 150; if (j0 < 0) j0 = 0;
    int j1 = 2 * t + 151; if (j1 > 1024) j1 = 1024;
    for (int jj = j0; jj < j1; ++jj) {
        float d = step * (float)jj - ft;
        acc = fmaf(__expf(d * d * (-0.02f)),        // -1/(2*sigma^2), sigma=5
                   wca3ca1[(l << 20) + jj * 1024 + c], acc);
    }
    sca3[idx] = sigmoidf_(10.0f * acc - 5.0f);
}

// -------------------------------------------------------------------------
// One-time: initialize recurrent state (f32 master + f16 shadow for ec3)
// -------------------------------------------------------------------------
__global__ __launch_bounds__(256) void init_kernel(
    const float* __restrict__ ec3_last, const float* __restrict__ ec5_last,
    float* __restrict__ ec3f, _Float16* __restrict__ ec3h,
    float* __restrict__ ec5f)
{
    unsigned idx = blockIdx.x * 256u + threadIdx.x;      // < 2*64*1024
    if (idx >= 131072u) return;
    float v = ec3_last[idx];
    ec3f[idx] = v;
    ec3h[idx] = (_Float16)v;
    ec5f[idx] = ec5_last[idx];
}

// -------------------------------------------------------------------------
// Core: one wave computes TWO vertically adjacent 16x16 tiles (mT0, mT0+1)
// for one nTile, sharing a single B operand per K-chunk: one 32B B load
// feeds two independent v_wmma_f32_16x16x32_f16 issues (2x matrix-pipe ILP,
// 48B loaded per WMMA instead of 64B).
// -------------------------------------------------------------------------
__device__ __forceinline__ void gemm_tile2(const _Float16* __restrict__ A,
                                           const _Float16* __restrict__ Bp,
                                           int mT0, int nTile, int lane,
                                           v8f& acc0, v8f& acc1)
{
    const int mrow = lane & 15;
    const int kbase = (lane >> 4) * 8;
    const _Float16* arow0 = A + (mT0 * 16 + mrow) * 1024 + kbase;
    const _Float16* arow1 = arow0 + 16 * 1024;
    const _Float16* bp = Bp + (size_t)nTile * 32 * 512 + lane * 16;
    #pragma unroll 2
    for (int kk = 0; kk < 32; ++kk) {
        v16h a0, a1, b;
        ((v8h*)&a0)[0] = *(const v8h*)(arow0 + kk * 32);        // K 0..7  (+kbase)
        ((v8h*)&a0)[1] = *(const v8h*)(arow0 + kk * 32 + 16);   // K 16..23 (+kbase)
        ((v8h*)&a1)[0] = *(const v8h*)(arow1 + kk * 32);
        ((v8h*)&a1)[1] = *(const v8h*)(arow1 + kk * 32 + 16);
        b = *(const v16h*)(bp + kk * 512);
        __builtin_prefetch(bp + kk * 512 + 512, 0, 1);          // speculative; tail OOB dropped
        acc0 = __builtin_amdgcn_wmma_f32_16x16x32_f16(false, a0, false, b,
                                                      (short)0, acc0, false, false);
        acc1 = __builtin_amdgcn_wmma_f32_16x16x32_f16(false, a1, false, b,
                                                      (short)0, acc1, false, false);
    }
}

// -------------------------------------------------------------------------
// Per step, per layer: ca1 = relu(sca3 * (1 + 3*sig(ec3 @ W)) - bias)
// 256 tiles -> 128 waves (2 tiles/wave) -> 16 blocks x 8 waves
// -------------------------------------------------------------------------
__global__ __launch_bounds__(256) void ca1_kernel(
    const _Float16* __restrict__ ec3h, const _Float16* __restrict__ wpack,
    const float* __restrict__ sca3t, const float* __restrict__ bias,
    float* __restrict__ ca1f, _Float16* __restrict__ ca1h,
    float* __restrict__ his)
{
    int lane = threadIdx.x & 31;
    int wg = blockIdx.x * 8 + (threadIdx.x >> 5);   // [0,128)
    int nTile = wg >> 1;
    int mT0 = (wg & 1) * 2;
    v8f acc0 = {}, acc1 = {};
    gemm_tile2(ec3h, wpack, mT0, nTile, lane, acc0, acc1);
    int n = nTile * 16 + (lane & 15);
    int mhi = (lane >> 4) * 8;
    float s = sca3t[n];
    float bb = bias[n];
    #pragma unroll
    for (int q = 0; q < 2; ++q) {
        v8f& acc = q ? acc1 : acc0;
        #pragma unroll
        for (int r = 0; r < 8; ++r) {
            int m = (mT0 + q) * 16 + mhi + r;
            float val = fmaxf(s * (1.0f + 3.0f * sigmoidf_(acc[r])) - bb, 0.0f);
            ca1f[m * 1024 + n] = val;
            ca1h[m * 1024 + n] = (_Float16)val;
            if (m == 0) his[n] = val;
        }
    }
}

// -------------------------------------------------------------------------
// Per step, per layer: two GEMMs sharing A = ca1 (f16):
//   half 0 (waves 0..127):   u = ca1 @ wca1ec5 -> ec5/ec3 update + history
//   half 1 (waves 128..255): v = ca1 @ inter_wT -> next layer's ec3input
// half 1 only launched/written for layer 0 (avoids race on ec3in buffer).
// -------------------------------------------------------------------------
__global__ __launch_bounds__(256) void ec5ec3_kernel(
    const _Float16* __restrict__ ca1h,
    const _Float16* __restrict__ wu, const _Float16* __restrict__ wv,
    const float* __restrict__ ec5bias, const float* __restrict__ interb,
    const float* __restrict__ xin, long xin_stride,
    float* __restrict__ ec5f, float* __restrict__ ec3f,
    _Float16* __restrict__ ec3h, float* __restrict__ ec3in_out,
    float* __restrict__ his_ec3, float* __restrict__ his_ec5,
    int write_inter)
{
    int lane = threadIdx.x & 31;
    int wgid = blockIdx.x * 8 + (threadIdx.x >> 5);  // [0,256)
    int hsel = wgid >> 7;
    int sub = wgid & 127;
    int nTile = sub >> 1;
    int mT0 = (sub & 1) * 2;
    const _Float16* Bp = hsel ? wv : wu;
    v8f acc0 = {}, acc1 = {};
    gemm_tile2(ca1h, Bp, mT0, nTile, lane, acc0, acc1);
    int n = nTile * 16 + (lane & 15);
    int mhi = (lane >> 4) * 8;
    if (hsel == 0) {
        float eb = ec5bias[n];
        #pragma unroll
        for (int q = 0; q < 2; ++q) {
            v8f& acc = q ? acc1 : acc0;
            #pragma unroll
            for (int r = 0; r < 8; ++r) {
                int m = (mT0 + q) * 16 + mhi + r;
                float e  = ec5f[m * 1024 + n] + acc[r] + eb;   // 10*TS == 1.0
                float e5 = 0.69f + 0.3f * sigmoidf_(4.0f * (e - 0.3f));
                float e3 = e5 * ec3f[m * 1024 + n] + 0.6f * xin[(long)m * xin_stride + n];
                ec5f[m * 1024 + n] = e5;
                ec3f[m * 1024 + n] = e3;
                ec3h[m * 1024 + n] = (_Float16)e3;
                if (m == 0) { his_ec5[n] = e5; his_ec3[n] = e3; }
            }
        }
    } else if (write_inter) {
        float ib = interb[n];
        #pragma unroll
        for (int q = 0; q < 2; ++q) {
            v8f& acc = q ? acc1 : acc0;
            #pragma unroll
            for (int r = 0; r < 8; ++r) {
                int m = (mT0 + q) * 16 + mhi + r;
                ec3in_out[m * 1024 + n] = acc[r] + ib;
            }
        }
    }
}

// -------------------------------------------------------------------------
// Final: actCell = ca1[L-1] @ wca1act + actbias   (64x1024 x 1024x2)
// -------------------------------------------------------------------------
__global__ void act_kernel(const float* __restrict__ ca1f1,
                           const float* __restrict__ wact,
                           const float* __restrict__ ab,
                           float* __restrict__ out)
{
    int tid = threadIdx.x;
    if (tid >= 128) return;
    int m = tid >> 1, a = tid & 1;
    float acc = ab[a];
    for (int k = 0; k < 1024; ++k)
        acc = fmaf(ca1f1[m * 1024 + k], wact[k * 2 + a], acc);
    out[m * 2 + a] = acc;
}

extern "C" void kernel_launch(void* const* d_in, const int* in_sizes, int n_in,
                              void* d_out, int out_size, void* d_ws, size_t ws_size,
                              hipStream_t stream) {
    (void)in_sizes; (void)n_in; (void)out_size; (void)ws_size;
    const float* cue      = (const float*)d_in[0];   // (BS,T,EC)
    const float* ec3_last = (const float*)d_in[1];   // (L,BS,EC)
    const float* ec5_last = (const float*)d_in[2];   // (L,BS,EC)
    /* d_in[3] ca1_last: carry-only, never observed in outputs */
    const float* wca3ca1  = (const float*)d_in[4];   // (L,CA3,CA1)
    const float* wec3ca1  = (const float*)d_in[5];   // (L,EC,CA1)
    const float* wca1ec5  = (const float*)d_in[6];   // (L,CA1,EC)
    const float* ca1bias  = (const float*)d_in[7];   // (L,CA1)
    const float* ec5bias  = (const float*)d_in[8];   // (L,EC)
    const float* inter_w  = (const float*)d_in[9];   // (L,EC,CA1)
    const float* inter_b  = (const float*)d_in[10];  // (L,EC)
    const float* wca1act  = (const float*)d_in[11];  // (CA1,ACT)
    const float* actbias  = (const float*)d_in[12];  // (ACT,)
    float* out = (float*)d_out;

    // Workspace carve-up (~19.1 MB)
    char* base = (char*)d_ws;
    size_t off = 0;
    auto ALLOC = [&](size_t bytes) {
        char* p = base + off;
        off += (bytes + 255) & ~(size_t)255;
        return p;
    };
    _Float16* packW = (_Float16*)ALLOC(6291456ull * 2);  // 3 mats x L x 1M f16
    float*    sca3  = (float*)   ALLOC(1048576ull * 4);  // (T,L,CA1)
    float*    ec3f  = (float*)   ALLOC(131072ull * 4);   // (L,BS,EC)
    float*    ec5f  = (float*)   ALLOC(131072ull * 4);
    float*    ca1f  = (float*)   ALLOC(131072ull * 4);
    _Float16* ec3h  = (_Float16*)ALLOC(131072ull * 2);
    _Float16* ca1h  = (_Float16*)ALLOC(131072ull * 2);
    float*    ec3in = (float*)   ALLOC(65536ull * 4);    // (BS,EC) inter-layer

    // Hoisted work
    pack_kernel<<<24576, 256, 0, stream>>>(wec3ca1, wca1ec5, inter_w, packW);
    sca3_kernel<<<4096, 256, 0, stream>>>(wca3ca1, sca3);
    init_kernel<<<512, 256, 0, stream>>>(ec3_last, ec5_last, ec3f, ec3h, ec5f);

    // Output layout: actCell[128] | ec3his[L*T*EC] | ec5his | ca1his
    float* ec3his = out + 128;
    float* ec5his = out + 128 + 1048576;
    float* ca1his = out + 128 + 2097152;

    for (int t = 0; t < T_; ++t) {
        for (int l = 0; l < L_; ++l) {
            ca1_kernel<<<16, 256, 0, stream>>>(
                ec3h + l * 65536,
                packW + (size_t)(0 * 2 + l) * 1048576,
                sca3 + (size_t)(t * 2 + l) * 1024,
                ca1bias + l * 1024,
                ca1f + l * 65536, ca1h + l * 65536,
                ca1his + (size_t)l * 524288 + (size_t)t * 1024);
            int nb = (l == 0) ? 32 : 16;   // inter-GEMM only needed for layer 0
            ec5ec3_kernel<<<nb, 256, 0, stream>>>(
                ca1h + l * 65536,
                packW + (size_t)(1 * 2 + l) * 1048576,
                packW + (size_t)(2 * 2 + l) * 1048576,
                ec5bias + l * 1024, inter_b + l * 1024,
                (l == 0) ? (cue + (size_t)t * 1024) : ec3in,
                (l == 0) ? (long)(T_ * N_) : (long)N_,
                ec5f + l * 65536, ec3f + l * 65536, ec3h + l * 65536,
                ec3in,
                ec3his + (size_t)l * 524288 + (size_t)t * 1024,
                ec5his + (size_t)l * 524288 + (size_t)t * 1024,
                (l == 0) ? 1 : 0);
        }
    }

    act_kernel<<<1, 128, 0, stream>>>(ca1f + 65536, wca1act, actbias, out);
}
